// FasterRCNN_17987323036104
// MI455X (gfx1250) — matrix operand order repaired and checked by reading, not compiled, not run on hardware
//
#include <hip/hip_runtime.h>
#include <cfloat>

// ---------------------------------------------------------------------------
// Faster-RCNN head for MI455X (gfx1250, wave32).
//
// Roofline: FC GEMM = 26.3 GFLOP, fc_w = 205 MB -> HBM floor 8.8 us needs
// ~3 PFLOPS. f32 WMMA (K=4) tops out ~0.6 PFLOPS -> compute bound. The 16-bit
// WMMA shape (K=32) gives 8x the FLOPs per issue slot, so:
//   K0: fc_w f32 -> f16 streaming convert (f16 copy stays hot in 192MB L2)
//   K1: ROI adaptive max-pool, emits f16 pooled [256,50176]
//   K2: GEMM via V_WMMA_F32_16X16X32_F16, split-K=8 (256 WGs / 2048 waves),
//       f32 partials (deterministic fixed-order reduce, no float atomics)
//   K3: reduce partials + bias + ReLU -> r [256,1024] f32
//   K4: cls/reg heads + argmax gather -> d_out
// Workspace required: fcw_h 102.8MB + pooled_h 25.7MB + part 8MB + r 1MB
//                     = 137,887,744 bytes (~131.5 MiB).
// ---------------------------------------------------------------------------

typedef _Float16 v8h  __attribute__((ext_vector_type(8)));
typedef _Float16 v16h __attribute__((ext_vector_type(16)));
typedef float    v8f  __attribute__((ext_vector_type(8)));

#define CCH    1024
#define FH     100
#define FW     50
#define POOL   7
#define NROI   256
#define NCLS   21
#define KDIM   (CCH * POOL * POOL)     // 50176
#define NOUT   1024
#define SPLITK 8
#define KCHUNK (KDIM / SPLITK)         // 6272 (divisible by 32)
#define FCW_ELTS ((size_t)NOUT * KDIM) // 51,380,224

// ---------------------------------------------------------------------------
// Kernel 0: fc_w f32 -> f16, pure streaming (8 elements / thread, b128 I/O).
// ---------------------------------------------------------------------------
__global__ __launch_bounds__(256) void cvt_fcw_kernel(
    const float* __restrict__ src, _Float16* __restrict__ dst)
{
  const size_t i = ((size_t)blockIdx.x * 256 + threadIdx.x) * 8;  // grid covers exactly FCW_ELTS
  const float4 a = *reinterpret_cast<const float4*>(src + i);
  const float4 b = *reinterpret_cast<const float4*>(src + i + 4);
  v8h o;
  o[0] = (_Float16)a.x; o[1] = (_Float16)a.y;
  o[2] = (_Float16)a.z; o[3] = (_Float16)a.w;
  o[4] = (_Float16)b.x; o[5] = (_Float16)b.y;
  o[6] = (_Float16)b.z; o[7] = (_Float16)b.w;
  *reinterpret_cast<v8h*>(dst + i) = o;
}

// ---------------------------------------------------------------------------
// Kernel 1: ROI adaptive max pooling -> f16.
// One block per ROI; thread owns (channel, bin) outputs; bounds recomputed
// inline (no dynamically indexed private arrays -> no scratch).
// Reference quirk preserved: (h0,w0,h1,w1) = (x1,y1,x2,y2)//16.
// ---------------------------------------------------------------------------
__global__ __launch_bounds__(256) void roi_pool_kernel(
    const float*  __restrict__ fm,      // [1024,100,50]
    const int*    __restrict__ prop,    // [256,4]
    _Float16*     __restrict__ pooled)  // [256,50176] f16
{
  const int roi = blockIdx.x;
  const int4 p = reinterpret_cast<const int4*>(prop)[roi];
  const int h0 = (p.x > 0 ? p.x : 0) >> 4;
  const int w0 = (p.y > 0 ? p.y : 0) >> 4;
  const int h1 = (p.z > 0 ? p.z : 0) >> 4;
  const int w1 = (p.w > 0 ? p.w : 0) >> 4;
  const int sh = h1 - h0 + 1;
  const int sw = w1 - w0 + 1;

  _Float16* out = pooled + (size_t)roi * KDIM;
  for (int idx = threadIdx.x; idx < KDIM; idx += 256) {
    const int c  = idx / (POOL * POOL);
    const int b  = idx - c * (POOL * POOL);
    const int bi = b / POOL;
    const int bj = b - bi * POOL;
    const int hsv = h0 + (bi * sh) / POOL;
    const int hev = h0 + ((bi + 1) * sh + POOL - 1) / POOL;
    const int wsv = w0 + (bj * sw) / POOL;
    const int wev = w0 + ((bj + 1) * sw + POOL - 1) / POOL;

    const float* base = fm + (size_t)c * (FH * FW);
    float m = -FLT_MAX;
    for (int h = hsv; h < hev; ++h) {
      const float* row = base + h * FW;
      for (int w = wsv; w < wev; ++w)
        m = fmaxf(m, row[w]);
    }
    out[idx] = (_Float16)m;   // [roi][c*49 + bi*7 + bj] matches fc_w K order
  }
}

// ---------------------------------------------------------------------------
// Kernel 2: split-K GEMM partials via V_WMMA_F32_16X16X32_F16.
//
// 16-bit fragment layouts used here (contiguous k; h = lane/16, lid = lane%16):
//   A 16x32 : M = lid; elems 0-7  -> K = k + 16h + 0..7
//                      elems 8-15 -> K = k + 16h + 8..15   (two b128 loads)
//   B 32x16 : N = lid; elems 0-15 -> K = k + 16h + 0..15      (two b128 loads)
//   C/D     : N = lid; elem v     -> M = v + 8h
// Wave tile 16(M)x64(N) (4 accums); 8 waves stacked in M -> 128x64 block.
// Grid (2, 16, 8): M-blocks x N-blocks x K-splits = 256 workgroups.
// ---------------------------------------------------------------------------
__global__ __launch_bounds__(256) void fc_wmma_kernel(
    const _Float16* __restrict__ pooled, // [256, 50176] f16
    const _Float16* __restrict__ fcw,    // [1024, 50176] f16
    float*          __restrict__ part)   // [8][256][1024] f32 partials
{
  const int lane = threadIdx.x & 31;
  const int wave = threadIdx.x >> 5;
  const int half = lane >> 4;
  const int lid  = lane & 15;

  const int m0 = (blockIdx.x * 8 + wave) * 16;   // gridDim.x = 2
  const int n0 = blockIdx.y * 64;                // gridDim.y = 16
  const int k0 = blockIdx.z * KCHUNK;            // gridDim.z = 8

  const _Float16* pA  = pooled + (size_t)(m0 + lid) * KDIM + k0 + 16 * half;
  const _Float16* pB0 = fcw + (size_t)(n0 +  0 + lid) * KDIM + k0 + 16 * half;
  const _Float16* pB1 = fcw + (size_t)(n0 + 16 + lid) * KDIM + k0 + 16 * half;
  const _Float16* pB2 = fcw + (size_t)(n0 + 32 + lid) * KDIM + k0 + 16 * half;
  const _Float16* pB3 = fcw + (size_t)(n0 + 48 + lid) * KDIM + k0 + 16 * half;

  v8f acc0 = {}, acc1 = {}, acc2 = {}, acc3 = {};

#pragma unroll 2
  for (int k = 0; k < KCHUNK; k += 32) {
    const v8h aL = *reinterpret_cast<const v8h*>(pA + k);
    const v8h aH = *reinterpret_cast<const v8h*>(pA + k + 8);
    const v16h a = __builtin_shufflevector(aL, aH, 0,1,2,3,4,5,6,7,
                                                   8,9,10,11,12,13,14,15);
    const v8h b0L = *reinterpret_cast<const v8h*>(pB0 + k);
    const v8h b0H = *reinterpret_cast<const v8h*>(pB0 + k + 8);
    const v16h b0 = __builtin_shufflevector(b0L, b0H, 0,1,2,3,4,5,6,7,
                                                      8,9,10,11,12,13,14,15);
    const v8h b1L = *reinterpret_cast<const v8h*>(pB1 + k);
    const v8h b1H = *reinterpret_cast<const v8h*>(pB1 + k + 8);
    const v16h b1 = __builtin_shufflevector(b1L, b1H, 0,1,2,3,4,5,6,7,
                                                      8,9,10,11,12,13,14,15);
    const v8h b2L = *reinterpret_cast<const v8h*>(pB2 + k);
    const v8h b2H = *reinterpret_cast<const v8h*>(pB2 + k + 8);
    const v16h b2 = __builtin_shufflevector(b2L, b2H, 0,1,2,3,4,5,6,7,
                                                      8,9,10,11,12,13,14,15);
    const v8h b3L = *reinterpret_cast<const v8h*>(pB3 + k);
    const v8h b3H = *reinterpret_cast<const v8h*>(pB3 + k + 8);
    const v16h b3 = __builtin_shufflevector(b3L, b3H, 0,1,2,3,4,5,6,7,
                                                      8,9,10,11,12,13,14,15);

    acc0 = __builtin_amdgcn_wmma_f32_16x16x32_f16(false, a, false, b0,
                                                  (short)0, acc0, false, false);
    acc1 = __builtin_amdgcn_wmma_f32_16x16x32_f16(false, a, false, b1,
                                                  (short)0, acc1, false, false);
    acc2 = __builtin_amdgcn_wmma_f32_16x16x32_f16(false, a, false, b2,
                                                  (short)0, acc2, false, false);
    acc3 = __builtin_amdgcn_wmma_f32_16x16x32_f16(false, a, false, b3,
                                                  (short)0, acc3, false, false);
  }

  float* pOut = part + (size_t)blockIdx.z * (NROI * NOUT);
#pragma unroll
  for (int v = 0; v < 8; ++v) {
    const int m = m0 + v + 8 * half;
    float* rr = pOut + (size_t)m * NOUT + n0 + lid;
    rr[ 0] = acc0[v];
    rr[16] = acc1[v];
    rr[32] = acc2[v];
    rr[48] = acc3[v];
  }
}

// ---------------------------------------------------------------------------
// Kernel 3: deterministic split-K reduce + bias + ReLU -> r [256,1024] f32.
// ---------------------------------------------------------------------------
__global__ __launch_bounds__(256) void reduce_kernel(
    const float* __restrict__ part,  // [8][256][1024]
    const float* __restrict__ fcb,   // [1024]
    float*       __restrict__ r)     // [256,1024]
{
  const int idx = blockIdx.x * 256 + threadIdx.x;   // grid = 1024 -> 262144
  float s = fcb[idx & (NOUT - 1)];
#pragma unroll
  for (int z = 0; z < SPLITK; ++z)
    s += part[(size_t)z * (NROI * NOUT) + idx];
  r[idx] = fmaxf(s, 0.0f);
}

// ---------------------------------------------------------------------------
// Kernel 4: cls/reg heads + per-ROI argmax gather.
// ---------------------------------------------------------------------------
__global__ __launch_bounds__(128) void heads_kernel(
    const float* __restrict__ r,       // [256,1024]
    const float* __restrict__ clsw,    // [21,1024]
    const float* __restrict__ clsb,    // [21]
    const float* __restrict__ regw,    // [84,1024]
    const float* __restrict__ regb,    // [84]
    float*       __restrict__ cls_out, // [256,21]
    float*       __restrict__ reg_out) // [256,4]
{
  __shared__ float rs[NOUT];
  __shared__ float cls_s[NCLS];
  __shared__ float reg_s[NCLS * 4];

  const int roi = blockIdx.x;
  for (int k = threadIdx.x; k < NOUT; k += 128)
    rs[k] = r[(size_t)roi * NOUT + k];
  __syncthreads();

  const int j = threadIdx.x;
  if (j < NCLS + NCLS * 4) {
    const float* w;
    float acc;
    if (j < NCLS) { w = clsw + (size_t)j * NOUT;          acc = clsb[j]; }
    else          { w = regw + (size_t)(j - NCLS) * NOUT; acc = regb[j - NCLS]; }
#pragma unroll 8
    for (int k = 0; k < NOUT; ++k)
      acc += rs[k] * w[k];
    if (j < NCLS) { cls_out[roi * NCLS + j] = acc; cls_s[j] = acc; }
    else          { reg_s[j - NCLS] = acc; }
  }
  __syncthreads();

  if (threadIdx.x == 0) {
    int best = 0;
    float bv = cls_s[0];
    for (int t = 1; t < NCLS; ++t)       // strict '>' => first max (jnp.argmax)
      if (cls_s[t] > bv) { bv = cls_s[t]; best = t; }
#pragma unroll
    for (int q = 0; q < 4; ++q)
      reg_out[roi * 4 + q] = reg_s[best * 4 + q];
  }
}

// ---------------------------------------------------------------------------
extern "C" void kernel_launch(void* const* d_in, const int* in_sizes, int n_in,
                              void* d_out, int out_size, void* d_ws, size_t ws_size,
                              hipStream_t stream) {
  const float* fm   = (const float*)d_in[0];  // feature_map [1024,100,50]
  const int*   prop = (const int*)  d_in[1];  // proposals   [256,4]
  const float* fcw  = (const float*)d_in[2];  // fc_w  [1024,50176]
  const float* fcb  = (const float*)d_in[3];  // fc_b  [1024]
  const float* clsw = (const float*)d_in[4];  // cls_w [21,1024]
  const float* clsb = (const float*)d_in[5];  // cls_b [21]
  const float* regw = (const float*)d_in[6];  // reg_w [84,1024]
  const float* regb = (const float*)d_in[7];  // reg_b [84]

  float* out = (float*)d_out;                 // cls [256*21] ++ reg [256*4]

  // Workspace carve-up (needs ~131.5 MiB).
  char* ws = (char*)d_ws;
  _Float16* fcw_h   = (_Float16*)ws;                       // 102,760,448 B
  _Float16* pooled  = (_Float16*)(ws + FCW_ELTS * sizeof(_Float16));
  float*    part    = (float*)((char*)pooled + (size_t)NROI * KDIM * sizeof(_Float16));
  float*    rbuf    = (float*)((char*)part + (size_t)SPLITK * NROI * NOUT * sizeof(float));

  // K0: fc_w -> f16 (51,380,224 elements, 8 per thread, 256 threads/block)
  cvt_fcw_kernel<<<FCW_ELTS / (256 * 8), 256, 0, stream>>>(fcw, fcw_h);

  // K1: ROI max-pool -> f16 pooled
  roi_pool_kernel<<<NROI, 256, 0, stream>>>(fm, prop, pooled);

  // K2: WMMA f16 GEMM partials, grid (M/128, N/64, SPLITK)
  dim3 g2(2, 16, SPLITK);
  fc_wmma_kernel<<<g2, 256, 0, stream>>>(pooled, fcw_h, part);

  // K3: reduce + bias + ReLU
  reduce_kernel<<<(NROI * NOUT) / 256, 256, 0, stream>>>(part, fcb, rbuf);

  // K4: heads + argmax gather
  heads_kernel<<<NROI, 128, 0, stream>>>(rbuf, clsw, clsb, regw, regb,
                                         out, out + NROI * NCLS);
}
